// PredicateLevelEncoder_6777458393259
// MI455X (gfx1250) — matrix operand order repaired and checked by reading, not compiled, not run on hardware
//
#include <hip/hip_runtime.h>
#include <hip/hip_bf16.h>
#include <math.h>

typedef __attribute__((ext_vector_type(16))) _Float16 v16h;
typedef __attribute__((ext_vector_type(8)))  _Float16 v8h;
typedef __attribute__((ext_vector_type(8)))  float    v8f;
typedef __attribute__((ext_vector_type(4)))  float    f32x4;

// ---------------------------------------------------------------------------
// WMMA fragment loaders (CDNA5 16x16x32 f16 layouts, ISA 7.12.2).
// All loads are unconditional vector loads (addresses 16B-aligned: every K
// here is a multiple of 16 floats). For N-edge tiles the OOB column's row
// pointer is clamped to row 0 (keeps the load in-bounds); the resulting
// garbage lives only in D columns that the guarded store discards (B column n
// feeds only D column n, stored by the same lanes), so no zero-select needed.
// ---------------------------------------------------------------------------

// A tile 16x32 (MxK) from fp32 row-major: lane half selects K range,
// row = lane & 15. Fragment elements are K-contiguous: a[0..7] = ap[0..7],
// a[8..15] = ap[16..23] with ap at (row, k0 + half*8).
__device__ __forceinline__ v16h load_a_f32(const float* __restrict__ A,
                                           int row0, int k0, int ldA, int lane) {
  const int half = lane >> 4;
  const int m    = lane & 15;
  const float* ap = A + (size_t)(row0 + m) * ldA + k0 + half * 8;
  f32x4 x0 = *(const f32x4*)(ap);
  f32x4 x1 = *(const f32x4*)(ap + 4);
  f32x4 x2 = *(const f32x4*)(ap + 16);
  f32x4 x3 = *(const f32x4*)(ap + 20);
  v16h a;
#pragma unroll
  for (int i = 0; i < 4; ++i) {
    a[i]      = (_Float16)x0[i];
    a[4 + i]  = (_Float16)x1[i];
    a[8 + i]  = (_Float16)x2[i];
    a[12 + i] = (_Float16)x3[i];
  }
  return a;
}

// A tile from f16 LDS: two 16B ds_load_b128 per fragment.
__device__ __forceinline__ v16h load_a_lds(const _Float16* __restrict__ A,
                                           int row0, int k0, int ldA, int lane) {
  const int half = lane >> 4;
  const int m    = lane & 15;
  const _Float16* ap = A + (size_t)(row0 + m) * ldA + k0 + half * 8;
  v8h lo = *(const v8h*)(ap);
  v8h hi = *(const v8h*)(ap + 16);
  v16h a;
#pragma unroll
  for (int i = 0; i < 8; ++i) { a[i] = lo[i]; a[8 + i] = hi[i]; }
  return a;
}

// B tile 32x16 (KxN) where B[k][n] = W[col0+n][k], W row-major [N,K].
// lane half selects K range; col = lane & 15; b[0..7] = wp[0..7], b[8..15] = wp[16..23].
// OOB columns clamp to row 0 (garbage confined to unstored D columns).
__device__ __forceinline__ v16h load_bT_f32(const float* __restrict__ W,
                                            int col0, int k0, int ldW, int Nlim,
                                            int lane) {
  const int half = lane >> 4;
  const int n    = lane & 15;
  const int ng   = col0 + n;
  const int ngc  = (ng < Nlim) ? ng : 0;
  const float* wp = W + (size_t)ngc * ldW + k0 + half * 8;
  f32x4 x0 = *(const f32x4*)(wp);
  f32x4 x1 = *(const f32x4*)(wp + 4);
  f32x4 x2 = *(const f32x4*)(wp + 16);
  f32x4 x3 = *(const f32x4*)(wp + 20);
  v16h b;
#pragma unroll
  for (int i = 0; i < 4; ++i) {
    b[i]      = (_Float16)x0[i];
    b[4 + i]  = (_Float16)x1[i];
    b[8 + i]  = (_Float16)x2[i];
    b[12 + i] = (_Float16)x3[i];
  }
  return b;
}

// ---------------------------------------------------------------------------
// Generic WMMA GEMM: Y[M,N] = X[M,K] @ W[N,K]^T + b1 + b2
// Block = 128 threads (4 waves); each wave owns TWO 16x16 tiles sharing one
// A fragment. Grid = (ceil(N/128), M/16). M mult of 16, K mult of 32.
// ---------------------------------------------------------------------------
__global__ __launch_bounds__(128) void gemm_bias_wmma(
    const float* __restrict__ X, const float* __restrict__ W,
    const float* __restrict__ b1, const float* __restrict__ b2,
    float* __restrict__ Y, int M, int N, int K) {
  const int lane = threadIdx.x & 31;
  const int wave = threadIdx.x >> 5;
  const int row0 = blockIdx.y * 16;
  const int col0 = blockIdx.x * 128 + wave * 32;   // tile pair: col0, col0+16
  if (col0 >= N) return;  // wave-uniform: EXEC stays all-ones for WMMA

  v8f acc0 = {};
  v8f acc1 = {};
  for (int k0 = 0; k0 < K; k0 += 32) {
    v16h a  = load_a_f32(X, row0, k0, K, lane);
    v16h b0 = load_bT_f32(W, col0,      k0, K, N, lane);
    v16h b1f= load_bT_f32(W, col0 + 16, k0, K, N, lane);
    acc0 = __builtin_amdgcn_wmma_f32_16x16x32_f16(false, a, false, b0,
                                                  (short)0, acc0, false, false);
    acc1 = __builtin_amdgcn_wmma_f32_16x16x32_f16(false, a, false, b1f,
                                                  (short)0, acc1, false, false);
  }

  const int n     = lane & 15;
  const int mBase = (lane >> 4) * 8;
#pragma unroll 2
  for (int tt = 0; tt < 2; ++tt) {
    const int ng = col0 + tt * 16 + n;
    if (ng < N) {
      const float bias = (b1 ? b1[ng] : 0.f) + (b2 ? b2[ng] : 0.f);
      const v8f& acc = tt ? acc1 : acc0;
#pragma unroll
      for (int v = 0; v < 8; ++v)
        Y[(size_t)(row0 + mBase + v) * N + ng] = acc[v] + bias;
    }
  }
}

// ---------------------------------------------------------------------------
// Fused additive attention: one block per (b,t); B=32,T=64,N=36,H=512.
// scores_n = w . tanh(Wf[b,t]+Uo[b,n]+b_attn);  softmax over n;
// attn_obj = sum_n p_n * (Wf+Uo+b_attn).  Writes feats = [f3d | attn_obj].
// ---------------------------------------------------------------------------
__global__ __launch_bounds__(256) void attn_fuse(
    const float* __restrict__ f3d, const float* __restrict__ Wf,
    const float* __restrict__ Uo, const float* __restrict__ b_attn,
    const float* __restrict__ wv, const float* __restrict__ wb,
    float* __restrict__ feats) {
  const int row  = blockIdx.x;       // b*64 + t
  const int b    = row >> 6;
  const int tid  = threadIdx.x;
  const int lane = tid & 31;
  const int wave = tid >> 5;

  __shared__ float sWfb[512];
  __shared__ float sw[512];
  __shared__ float sP[40];

  for (int h = tid; h < 512; h += 256) {
    sWfb[h] = Wf[(size_t)row * 512 + h] + b_attn[h];
    sw[h]   = wv[h];
  }
  __syncthreads();

  const float* UoB = Uo + (size_t)b * 36 * 512;
  for (int n = wave; n < 36; n += 8) {     // one object per wave32
    const float* un = UoB + (size_t)n * 512;
    float p = 0.f;
    for (int h = lane; h < 512; h += 32)
      p += tanhf(sWfb[h] + un[h]) * sw[h];
#pragma unroll
    for (int off = 16; off > 0; off >>= 1)
      p += __shfl_xor(p, off, 32);
    if (lane == 0) sP[n] = p + wb[0];
  }
  __syncthreads();

  if (tid == 0) {  // softmax over 36 objects
    float mx = sP[0];
    for (int n = 1; n < 36; ++n) mx = fmaxf(mx, sP[n]);
    float s = 0.f;
    for (int n = 0; n < 36; ++n) { float e = expf(sP[n] - mx); sP[n] = e; s += e; }
    const float inv = 1.f / s;
    for (int n = 0; n < 36; ++n) sP[n] *= inv;
  }
  __syncthreads();

  for (int h = tid; h < 512; h += 256) {
    const float wfb = sWfb[h];
    float acc = 0.f;
#pragma unroll 4
    for (int n = 0; n < 36; ++n)
      acc += sP[n] * (wfb + UoB[(size_t)n * 512 + h]);
    feats[(size_t)row * 1024 + 512 + h] = acc;
    feats[(size_t)row * 1024 + h]       = f3d[(size_t)row * 512 + h];
  }
}

// ---------------------------------------------------------------------------
// BiLSTM recurrence. grid.x = direction (0 fwd, 1 bwd). 512 threads = 16 waves.
// Per step: g = Gpre[:,t] + h @ Whh^T via WMMA (M=32,N=1024,K=256): each wave
// owns 4 N-tiles per mi and reuses one A fragment across 4 WMMAs per k-step.
// h kept as f16 in LDS (A source via ds_load_b128), c f32 in LDS,
// gate buffer 128 KB in LDS (dynamic: 176 KB total of the 320 KB WGP pool).
// ---------------------------------------------------------------------------
__global__ __launch_bounds__(512) void bilstm_recur(
    const float* __restrict__ Gf, const float* __restrict__ Gb,
    const float* __restrict__ WhhF, const float* __restrict__ WhhB,
    float* __restrict__ out) {
  const int dir = blockIdx.x;
  const float* __restrict__ G   = dir ? Gb   : Gf;
  const float* __restrict__ Whh = dir ? WhhB : WhhF;
  const int tid  = threadIdx.x;
  const int lane = tid & 31;
  const int wave = tid >> 5;

  extern __shared__ char smem[];
  _Float16* h16  = (_Float16*)smem;              // 32*256 f16  = 16 KB
  float*    cst  = (float*)(smem + 16 * 1024);   // 32*256 f32  = 32 KB
  float*    gbuf = (float*)(smem + 48 * 1024);   // 32*1024 f32 = 128 KB

  for (int i = tid; i < 32 * 256; i += 512) { h16[i] = (_Float16)0.f; cst[i] = 0.f; }
  __syncthreads();

  const int colBase = wave * 64;                 // this wave's 4 N-tiles

  for (int step = 0; step < 64; ++step) {
    const int t = dir ? (63 - step) : step;

    // phase 1: gbuf = h @ Whh^T + Gpre[t]
#pragma unroll 1
    for (int mi = 0; mi < 2; ++mi) {
      v8f acc0 = {}, acc1 = {}, acc2 = {}, acc3 = {};
#pragma unroll 1
      for (int ks = 0; ks < 8; ++ks) {
        const int k0 = ks * 32;
        v16h a  = load_a_lds(h16, mi * 16, k0, 256, lane);
        v16h b0 = load_bT_f32(Whh, colBase,      k0, 256, 1024, lane);
        v16h b1 = load_bT_f32(Whh, colBase + 16, k0, 256, 1024, lane);
        v16h b2 = load_bT_f32(Whh, colBase + 32, k0, 256, 1024, lane);
        v16h b3 = load_bT_f32(Whh, colBase + 48, k0, 256, 1024, lane);
        acc0 = __builtin_amdgcn_wmma_f32_16x16x32_f16(false, a, false, b0,
                                                      (short)0, acc0, false, false);
        acc1 = __builtin_amdgcn_wmma_f32_16x16x32_f16(false, a, false, b1,
                                                      (short)0, acc1, false, false);
        acc2 = __builtin_amdgcn_wmma_f32_16x16x32_f16(false, a, false, b2,
                                                      (short)0, acc2, false, false);
        acc3 = __builtin_amdgcn_wmma_f32_16x16x32_f16(false, a, false, b3,
                                                      (short)0, acc3, false, false);
      }
      const int n  = lane & 15;
      const int mb = mi * 16 + (lane >> 4) * 8;
#pragma unroll 4
      for (int nt = 0; nt < 4; ++nt) {
        const v8f& acc = (nt == 0) ? acc0 : (nt == 1) ? acc1 : (nt == 2) ? acc2 : acc3;
        const int col = colBase + nt * 16 + n;
#pragma unroll
        for (int v = 0; v < 8; ++v) {
          const int bm = mb + v;
          gbuf[bm * 1024 + col] = acc[v] + G[((size_t)bm * 64 + t) * 1024 + col];
        }
      }
    }
    __syncthreads();

    // phase 2: gates (torch order i,f,g,o), update c/h, emit output
    for (int idx = tid; idx < 32 * 256; idx += 512) {
      const int bm = idx >> 8;
      const int hc = idx & 255;
      const float gi = gbuf[bm * 1024 + hc];
      const float gf = gbuf[bm * 1024 + 256 + hc];
      const float gg = gbuf[bm * 1024 + 512 + hc];
      const float go = gbuf[bm * 1024 + 768 + hc];
      const float si = 1.f / (1.f + expf(-gi));
      const float sf = 1.f / (1.f + expf(-gf));
      const float so = 1.f / (1.f + expf(-go));
      const float c  = sf * cst[idx] + si * tanhf(gg);
      const float h  = so * tanhf(c);
      cst[idx] = c;
      h16[idx] = (_Float16)h;
      out[((size_t)bm * 64 + t) * 512 + dir * 256 + hc] = h;
    }
    __syncthreads();
  }
}

// ---------------------------------------------------------------------------
// action[b,h] = max_t output[b,t,h]
// ---------------------------------------------------------------------------
__global__ void maxpool_t(const float* __restrict__ out, float* __restrict__ action) {
  const int idx = blockIdx.x * blockDim.x + threadIdx.x;
  if (idx >= 32 * 512) return;
  const int b = idx >> 9;
  const int h = idx & 511;
  const float* p = out + (size_t)b * 64 * 512 + h;
  float m = p[0];
  for (int t = 1; t < 64; ++t) m = fmaxf(m, p[(size_t)t * 512]);
  action[idx] = m;
}

// ---------------------------------------------------------------------------
extern "C" void kernel_launch(void* const* d_in, const int* in_sizes, int n_in,
                              void* d_out, int out_size, void* d_ws, size_t ws_size,
                              hipStream_t stream) {
  (void)in_sizes; (void)n_in; (void)out_size; (void)ws_size;

  const float* visual = (const float*)d_in[0];   // [32,64,2048]
  const float* objects = (const float*)d_in[1];  // [32,36,512]
  const float* lin_W  = (const float*)d_in[2];   // [512,2048]
  const float* lin_b  = (const float*)d_in[3];
  const float* W_W    = (const float*)d_in[4];   // [512,512]
  const float* W_b    = (const float*)d_in[5];
  const float* U_W    = (const float*)d_in[6];   // [512,512]
  const float* U_b    = (const float*)d_in[7];
  const float* b_attn = (const float*)d_in[8];   // [512]
  const float* w_W    = (const float*)d_in[9];   // [1,512]
  const float* w_b    = (const float*)d_in[10];  // [1]
  const float* Wih_f  = (const float*)d_in[11];  // [1024,1024]
  const float* Whh_f  = (const float*)d_in[12];  // [1024,256]
  const float* bih_f  = (const float*)d_in[13];
  const float* bhh_f  = (const float*)d_in[14];
  const float* Wih_b  = (const float*)d_in[15];
  const float* Whh_b  = (const float*)d_in[16];
  const float* bih_b  = (const float*)d_in[17];
  const float* bhh_b  = (const float*)d_in[18];
  const float* fc_W   = (const float*)d_in[19];  // [300,512]
  const float* fc_b   = (const float*)d_in[20];

  // workspace carve-out (floats), ~36 MB total
  float* ws     = (float*)d_ws;
  float* f3d    = ws;                  // 2048*512  = 1,048,576
  float* Wf     = f3d   + 1048576;     // 1,048,576
  float* Uo     = Wf    + 1048576;     // 1152*512  =   589,824
  float* feats  = Uo    + 589824;      // 2048*1024 = 2,097,152
  float* Gf     = feats + 2097152;     // 2,097,152
  float* Gb     = Gf    + 2097152;     // 2,097,152
  float* action = Gb    + 2097152;     // 32*512    =    16,384

  float* output = (float*)d_out;             // [32,64,512]
  float* asem   = output + 32 * 64 * 512;    // [32,300]

  // 1) f3d = visual @ lin_W^T + lin_b          (M=2048,N=512,K=2048)
  gemm_bias_wmma<<<dim3(4, 128), 128, 0, stream>>>(visual, lin_W, lin_b, nullptr,
                                                   f3d, 2048, 512, 2048);
  // 2) Wf = f3d @ W_W^T + W_b                  (M=2048,N=512,K=512)
  gemm_bias_wmma<<<dim3(4, 128), 128, 0, stream>>>(f3d, W_W, W_b, nullptr,
                                                   Wf, 2048, 512, 512);
  // 3) Uo = objects @ U_W^T + U_b              (M=1152,N=512,K=512)
  gemm_bias_wmma<<<dim3(4, 72), 128, 0, stream>>>(objects, U_W, U_b, nullptr,
                                                  Uo, 1152, 512, 512);
  // 4) fused attention -> feats [2048,1024]
  attn_fuse<<<2048, 256, 0, stream>>>(f3d, Wf, Uo, b_attn, w_W, w_b, feats);
  // 5) LSTM input gates: Gpre = feats @ Wih^T + (bih+bhh)   (M=2048,N=1024,K=1024)
  gemm_bias_wmma<<<dim3(8, 128), 128, 0, stream>>>(feats, Wih_f, bih_f, bhh_f,
                                                   Gf, 2048, 1024, 1024);
  gemm_bias_wmma<<<dim3(8, 128), 128, 0, stream>>>(feats, Wih_b, bih_b, bhh_b,
                                                   Gb, 2048, 1024, 1024);
  // 6) sequential BiLSTM recurrence (dir = blockIdx.x), 176 KB dynamic LDS
  bilstm_recur<<<2, 512, 176 * 1024, stream>>>(Gf, Gb, Whh_f, Whh_b, output);
  // 7) action = max over T
  maxpool_t<<<(32 * 512 + 255) / 256, 256, 0, stream>>>(output, action);
  // 8) action_semantics = action @ fc_W^T + fc_b   (M=32,N=300,K=512)
  gemm_bias_wmma<<<dim3(3, 2), 128, 0, stream>>>(action, fc_W, fc_b, nullptr,
                                                 asem, 32, 300, 512);
}
